// MultiHeadExternalAttention_8134668059232
// MI455X (gfx1250) — matrix operand order repaired and checked by reading, not compiled, not run on hardware
//
#include <hip/hip_runtime.h>
#include <hip/hip_bf16.h>
#include <stdint.h>

// ---------------------------------------------------------------------------
// MultiHeadExternalAttention, algebraically collapsed:
//   logits[tok, c] = x[tok, :512] @ W1[:512, c] + b1[c]      (c = h*16+m, 256 cols)
//   attn = L1_m( softmax_n( logits ) )
//   out[tok, e]   = attn[tok, :256] @ W2[:256, e] + b2[e]
// bf16 WMMA (v_wmma_f32_16x16x32_bf16) with f32 accumulate.
// B panels staged in 256KB LDS via async-to-LDS (ASYNCcnt); waves compute
// 64x64 output blocks (4x4 WMMA tiles) so 16 WMMAs run per wait.
// ---------------------------------------------------------------------------

#define EMBED   512
#define EXPAND  2048
#define HEADS   16
#define HDIM    128
#define MDIM    16
#define BATCH   32
#define SEQ     1024
#define MTOK    (BATCH * SEQ)   // 32768
#define C1      256             // HEADS * MDIM

typedef __bf16 v16bf __attribute__((ext_vector_type(16)));
typedef __bf16 v8bf  __attribute__((ext_vector_type(8)));
typedef float  v8f   __attribute__((ext_vector_type(8)));

// ---------------- swizzle helper: WMMA B-matrix (32x16 bf16 tile) layout ----
// lane L (0..31): column n = L&15, rows k = (L>>4)*16 + j, j=0..15
// flat index: ((kt*(N/16) + nt)*32 + lane)*16 + j
__device__ __forceinline__ size_t bswz_idx(int k, int n, int N) {
  int kt = k >> 5, nt = n >> 4;
  int rk = k & 31, rn = n & 15;
  int lane = ((rk >> 4) << 4) | rn;
  int j = rk & 15;
  return ((size_t)((kt * (N >> 4) + nt) * 32 + lane) << 4) + j;
}

// ---------------- fold W_in @ W_mk -> W1 (bf16, pre-swizzled), b1 -----------
__global__ void prep_w1_kernel(const float* __restrict__ W_in,
                               const float* __restrict__ b_in,
                               const float* __restrict__ W_mk,
                               const float* __restrict__ b_mk,
                               __hip_bfloat16* __restrict__ W1sw,
                               float* __restrict__ b1) {
  int idx = blockIdx.x * blockDim.x + threadIdx.x;
  if (idx < EMBED * C1) {
    int e = idx / C1, c = idx % C1;
    int h = c >> 4, m = c & 15;
    float s = 0.f;
    #pragma unroll 4
    for (int d = 0; d < HDIM; ++d)
      s += W_in[(size_t)e * EXPAND + h * HDIM + d] * W_mk[d * MDIM + m];
    W1sw[bswz_idx(e, c, C1)] = __float2bfloat16(s);
  }
  if (idx < C1) {
    int h = idx >> 4, m = idx & 15;
    float s = b_mk[m];
    for (int d = 0; d < HDIM; ++d)
      s += b_in[h * HDIM + d] * W_mk[d * MDIM + m];
    b1[idx] = s;
  }
}

// ---------------- fold W_mv @ W_out -> W2 (bf16, pre-swizzled), b2 ----------
__global__ void prep_w2_kernel(const float* __restrict__ W_mv,
                               const float* __restrict__ b_mv,
                               const float* __restrict__ W_out,
                               const float* __restrict__ b_out,
                               __hip_bfloat16* __restrict__ W2sw,
                               float* __restrict__ b2) {
  int idx = blockIdx.x * blockDim.x + threadIdx.x;
  if (idx < C1 * EMBED) {
    int c = idx / EMBED, e = idx % EMBED;
    int h = c >> 4, m = c & 15;
    float s = 0.f;
    #pragma unroll 4
    for (int d = 0; d < HDIM; ++d)
      s += W_mv[m * HDIM + d] * W_out[(size_t)(h * HDIM + d) * EMBED + e];
    W2sw[bswz_idx(c, e, EMBED)] = __float2bfloat16(s);
  }
  if (idx < EMBED) {
    float s = b_out[idx];
    for (int dd = 0; dd < EXPAND; ++dd)
      s += b_mv[dd & (HDIM - 1)] * W_out[(size_t)dd * EMBED + idx];
    b2[idx] = s;
  }
}

// ---------------- x f32 -> bf16 --------------------------------------------
__global__ void cvt_bf16_kernel(const float* __restrict__ x,
                                __hip_bfloat16* __restrict__ y, int n4) {
  int i = blockIdx.x * blockDim.x + threadIdx.x;
  if (i < n4) {
    float4 v = ((const float4*)x)[i];
    union { __hip_bfloat16 h[4]; uint2 u; } o;
    o.h[0] = __float2bfloat16(v.x);
    o.h[1] = __float2bfloat16(v.y);
    o.h[2] = __float2bfloat16(v.z);
    o.h[3] = __float2bfloat16(v.w);
    ((uint2*)y)[i] = o.u;
  }
}

// ---------------- bf16 WMMA GEMM:  C[M,N] = A[M,K] @ Bsw + bias ------------
// Block = 512 threads (16 waves). Whole swizzled B panel (K*N*2 = 256KB)
// staged in dynamic LDS via async-to-LDS. Each wave computes a 64x64 output
// block: 4 m-tiles x 4 n-tiles = 16 accumulators (128 VGPRs); per 32-deep
// K-step: 8 global b128 A-loads + 8 ds_load_b128 B-loads feed 16 WMMAs on
// distinct registers (no per-WMMA dscnt waits).
__global__ __launch_bounds__(512)
void gemm_bf16_wmma_kernel(const __hip_bfloat16* __restrict__ A,
                           const __hip_bfloat16* __restrict__ Bsw,
                           const float* __restrict__ bias,
                           float* __restrict__ C,
                           int M, int N, int K) {
  extern __shared__ __hip_bfloat16 ldsB[];

  // async panel load: global -> LDS, 16B per lane per op, ASYNCcnt-tracked
  {
    uint32_t lds_base = (uint32_t)(size_t)(void*)ldsB;
    int nchunk = (K * N) >> 3;            // 16B chunks
    for (int i = threadIdx.x; i < nchunk; i += blockDim.x) {
      uint32_t goff = (uint32_t)i << 4;
      asm volatile("global_load_async_to_lds_b128 %0, %1, %2"
                   :
                   : "v"(lds_base + goff), "v"(goff), "s"(Bsw)
                   : "memory");
    }
    asm volatile("s_wait_asynccnt 0" ::: "memory");
  }
  __syncthreads();

  const int lane = threadIdx.x & 31;
  const int wave = threadIdx.x >> 5;
  const int gw = blockIdx.x * (blockDim.x >> 5) + wave;
  const int nblocks = N >> 6;            // 64-wide n blocks
  const int mb = gw / nblocks;
  const int nb = gw % nblocks;
  if (mb * 64 >= M) return;              // wave-uniform: EXEC stays all-1s

  const int ntilesN = N >> 4;
  const int kbase = (lane >> 4) * 16;    // A-matrix lane layout: k = 16h + s (matches Bsw)
  const __hip_bfloat16* abase = A + (size_t)(mb * 64 + (lane & 15)) * K;

  v8f acc[4][4];
  const v8f vzero = {0.f, 0.f, 0.f, 0.f, 0.f, 0.f, 0.f, 0.f};
  #pragma unroll
  for (int i = 0; i < 4; ++i)
    #pragma unroll
    for (int j = 0; j < 4; ++j) acc[i][j] = vzero;

  const int ksteps = K >> 5;
  for (int kt = 0; kt < ksteps; ++kt) {
    // 4 A tiles: rows mb*64 + i*16 (+lane&15), constant offsets i*16*K
    union { v16bf v; v8bf h[2]; } au[4];
    #pragma unroll
    for (int i = 0; i < 4; ++i) {
      const __hip_bfloat16* ap = abase + (size_t)i * 16 * K + kt * 32 + kbase;
      au[i].h[0] = *(const v8bf*)(const void*)(ap);
      au[i].h[1] = *(const v8bf*)(const void*)(ap + 8);
    }
    if (kt + 1 < ksteps)
      __builtin_prefetch(abase + kt * 32 + 32, 0, 0);   // global_prefetch_b8

    // 4 B tiles from LDS (distinct registers)
    const __hip_bfloat16* bb =
        ldsB + (((size_t)(kt * ntilesN + nb * 4) * 32 + lane) << 4);
    v16bf breg[4];
    #pragma unroll
    for (int j = 0; j < 4; ++j)
      breg[j] = *(const v16bf*)(const void*)(bb + (j << 9));

    #pragma unroll
    for (int i = 0; i < 4; ++i)
      #pragma unroll
      for (int j = 0; j < 4; ++j)
        acc[i][j] = __builtin_amdgcn_wmma_f32_16x16x32_bf16(
            false, au[i].v, false, breg[j], (short)0, acc[i][j], false, false);
  }

  // epilogue: row = mb*64 + i*16 + (lane>>4)*8 + r, col = (nb*4+j)*16 + lane&15
  const int mhalf = (lane >> 4) * 8;
  #pragma unroll
  for (int i = 0; i < 4; ++i) {
    #pragma unroll
    for (int j = 0; j < 4; ++j) {
      int ncol = (nb * 4 + j) * 16 + (lane & 15);
      float bv = bias[ncol];
      float* cptr = C + (size_t)(mb * 64 + i * 16 + mhalf) * N + ncol;
      #pragma unroll
      for (int r = 0; r < 8; ++r)
        cptr[(size_t)r * N] = acc[i][j][r] + bv;
    }
  }
}

// ---------------- softmax over sequence dim + L1 over memory dim -----------
__global__ __launch_bounds__(256)
void softmax_l1_kernel(const float* __restrict__ Z,
                       __hip_bfloat16* __restrict__ attn) {
  const int b = blockIdx.x;
  const int c = threadIdx.x;
  const float* zp = Z + (size_t)b * SEQ * C1 + c;

  float mx = -3.402823466e38f, sm = 0.f;
  for (int n = 0; n < SEQ; ++n) {
    float v = zp[(size_t)n * C1];
    float m2 = fmaxf(mx, v);
    sm = sm * __expf(mx - m2) + __expf(v - m2);
    mx = m2;
  }
  const float inv = 1.0f / sm;

  __hip_bfloat16* op = attn + (size_t)b * SEQ * C1 + c;
  for (int n = 0; n < SEQ; ++n) {
    float s = __expf(zp[(size_t)n * C1] - mx) * inv;
    float g = s;                         // L1 sum over the 16 memory slots
    g += __shfl_xor(g, 8, 16);
    g += __shfl_xor(g, 4, 16);
    g += __shfl_xor(g, 2, 16);
    g += __shfl_xor(g, 1, 16);
    op[(size_t)n * C1] = __float2bfloat16(s / (g + 1e-9f));
  }
}

// ---------------------------------------------------------------------------
extern "C" void kernel_launch(void* const* d_in, const int* in_sizes, int n_in,
                              void* d_out, int out_size, void* d_ws, size_t ws_size,
                              hipStream_t stream) {
  const float* x     = (const float*)d_in[0];
  const float* W_in  = (const float*)d_in[1];
  const float* b_in  = (const float*)d_in[2];
  const float* W_mk  = (const float*)d_in[3];
  const float* b_mk  = (const float*)d_in[4];
  const float* W_mv  = (const float*)d_in[5];
  const float* b_mv  = (const float*)d_in[6];
  const float* W_out = (const float*)d_in[7];
  const float* b_out = (const float*)d_in[8];
  float* out = (float*)d_out;

  char* ws = (char*)d_ws;
  __hip_bfloat16* xbf  = (__hip_bfloat16*)(ws);              // 32 MB
  __hip_bfloat16* W1sw = (__hip_bfloat16*)(ws + 33554432);   // 256 KB
  float*          b1   = (float*)(ws + 33816576);            // 1 KB
  __hip_bfloat16* W2sw = (__hip_bfloat16*)(ws + 33817600);   // 256 KB
  float*          b2   = (float*)(ws + 34079744);            // 2 KB
  float*          Z    = (float*)(ws + 34081792);            // 32 MB
  __hip_bfloat16* attn = (__hip_bfloat16*)(ws + 67636224);   // 16 MB

  // 1) x -> bf16
  {
    int n4 = (MTOK * EMBED) / 4;
    cvt_bf16_kernel<<<(n4 + 255) / 256, 256, 0, stream>>>(x, xbf, n4);
  }
  // 2) fold weights (tiny)
  prep_w1_kernel<<<(EMBED * C1 + 255) / 256, 256, 0, stream>>>(
      W_in, b_in, W_mk, b_mk, W1sw, b1);
  prep_w2_kernel<<<(C1 * EMBED + 255) / 256, 256, 0, stream>>>(
      W_mv, b_mv, W_out, b_out, W2sw, b2);

  // 3) logits = xbf @ W1 + b1   [32768,512]x[512,256]
  {
    int waveunits = (MTOK / 64) * (C1 / 64);       // 512 * 4 = 2048
    int blocks = waveunits / 16;                   // 128
    gemm_bf16_wmma_kernel<<<blocks, 512, EMBED * C1 * 2, stream>>>(
        xbf, W1sw, b1, Z, MTOK, C1, EMBED);
  }
  // 4) softmax over n, L1 over m
  softmax_l1_kernel<<<BATCH, C1, 0, stream>>>(Z, attn);

  // 5) out = attn @ W2 + b2     [32768,256]x[256,512]
  {
    int waveunits = (MTOK / 64) * (EMBED / 64);    // 512 * 8 = 4096
    int blocks = waveunits / 16;                   // 256
    gemm_bf16_wmma_kernel<<<blocks, 512, C1 * EMBED * 2, stream>>>(
        attn, W2sw, b2, out, MTOK, EMBED, C1);
  }
}
